// EGNN_47528108097729
// MI455X (gfx1250) — matrix operand order repaired and checked by reading, not compiled, not run on hardware
//
#include <hip/hip_runtime.h>
#include <hip/hip_bf16.h>
#include <stdint.h>

#define N_NODES  10000
#define N_EDGES  160000
#define IN_DIM   512
#define HID      256
#define EDGE_DIM 51
#define K1       1088      // padded edge_in K (512 + 512 + 64)
#define K1SRC    1076
#define KN       768
#define TAILW    64

// packed-weight strides (bf16 elements)
#define WCHUNK   8192      // 16 ntiles * 32 lanes * 16 vals per K-chunk
#define WNT      512       // 32 lanes * 16 vals per n-tile

typedef __attribute__((ext_vector_type(16))) __bf16 v16bf;
typedef __attribute__((ext_vector_type(8)))  float  v8f;

struct Frag32B { uint4 lo, hi; };

// A-fragment (16-bit A 16x32 layout): lane holds row M=lane%16,
// K = 32*c + 16*(e/8) + 8*(lane/16) + e%8  -> two 16B chunks at +0 and +16 bf16.
__device__ inline v16bf load_afrag(const __bf16* base_kc, int lh) {
    const uint4* p = reinterpret_cast<const uint4*>(base_kc + lh * 8);
    Frag32B f; f.lo = p[0]; f.hi = p[2];
    return __builtin_bit_cast(v16bf, f);
}
// B-fragment: lane's 16 values contiguous (32B) at precomputed address.
__device__ inline v16bf load_b_at(const __bf16* p32) {
    const uint4* p = reinterpret_cast<const uint4*>(p32);
    Frag32B f; f.lo = p[0]; f.hi = p[1];
    return __builtin_bit_cast(v16bf, f);
}
__device__ inline v8f wmma_bf16(v16bf a, v16bf b, v8f c) {
    return __builtin_amdgcn_wmma_f32_16x16x32_bf16(false, a, false, b, (short)0, c, false, false);
}
__device__ inline float silu_f(float x) { return x / (1.0f + __expf(-x)); }
__device__ inline void zero8(v8f& v) {
#pragma unroll
    for (int j = 0; j < 8; ++j) v[j] = 0.0f;
}

// Pipelined GEMM over NC K-chunks, ONE row-tile (node kernel).
template<int NC>
__device__ inline void gemm_run(const __bf16* __restrict__ arow,
                                const __bf16* __restrict__ wb,
                                int lh, v8f acc[4]) {
    v16bf a  = load_afrag(arow, lh);
    v16bf b0 = load_b_at(wb + 0 * WNT);
    v16bf b1 = load_b_at(wb + 1 * WNT);
    v16bf b2 = load_b_at(wb + 2 * WNT);
    v16bf b3 = load_b_at(wb + 3 * WNT);
#pragma unroll
    for (int c = 0; c < NC; ++c) {
        v16bf an, bn0, bn1, bn2, bn3;
        if (c + 1 < NC) {
            an = load_afrag(arow + (c + 1) * 32, lh);
            const __bf16* w = wb + (size_t)(c + 1) * WCHUNK;
            bn0 = load_b_at(w + 0 * WNT);
            bn1 = load_b_at(w + 1 * WNT);
            bn2 = load_b_at(w + 2 * WNT);
            bn3 = load_b_at(w + 3 * WNT);
        }
        acc[0] = wmma_bf16(a, b0, acc[0]);
        acc[1] = wmma_bf16(a, b1, acc[1]);
        acc[2] = wmma_bf16(a, b2, acc[2]);
        acc[3] = wmma_bf16(a, b3, acc[3]);
        if (c + 1 < NC) { a = an; b0 = bn0; b1 = bn1; b2 = bn2; b3 = bn3; }
    }
}

// Pipelined GEMM over NC K-chunks, TWO row-tiles sharing the B-fragments
// (halves L2 B-traffic per output row). 8 WMMAs per chunk.
template<int NC>
__device__ inline void gemm_run2(const __bf16* __restrict__ a0row,
                                 const __bf16* __restrict__ a1row,
                                 const __bf16* __restrict__ wb,
                                 int lh, v8f acc[2][4]) {
    v16bf a0 = load_afrag(a0row, lh);
    v16bf a1 = load_afrag(a1row, lh);
    v16bf b0 = load_b_at(wb + 0 * WNT);
    v16bf b1 = load_b_at(wb + 1 * WNT);
    v16bf b2 = load_b_at(wb + 2 * WNT);
    v16bf b3 = load_b_at(wb + 3 * WNT);
#pragma unroll
    for (int c = 0; c < NC; ++c) {
        v16bf a0n, a1n, bn0, bn1, bn2, bn3;
        if (c + 1 < NC) {
            a0n = load_afrag(a0row + (c + 1) * 32, lh);
            a1n = load_afrag(a1row + (c + 1) * 32, lh);
            const __bf16* w = wb + (size_t)(c + 1) * WCHUNK;
            bn0 = load_b_at(w + 0 * WNT);
            bn1 = load_b_at(w + 1 * WNT);
            bn2 = load_b_at(w + 2 * WNT);
            bn3 = load_b_at(w + 3 * WNT);
        }
        acc[0][0] = wmma_bf16(a0, b0, acc[0][0]);
        acc[1][0] = wmma_bf16(a1, b0, acc[1][0]);
        acc[0][1] = wmma_bf16(a0, b1, acc[0][1]);
        acc[1][1] = wmma_bf16(a1, b1, acc[1][1]);
        acc[0][2] = wmma_bf16(a0, b2, acc[0][2]);
        acc[1][2] = wmma_bf16(a1, b2, acc[1][2]);
        acc[0][3] = wmma_bf16(a0, b3, acc[0][3]);
        acc[1][3] = wmma_bf16(a1, b3, acc[1][3]);
        if (c + 1 < NC) { a0 = a0n; a1 = a1n; b0 = bn0; b1 = bn1; b2 = bn2; b3 = bn3; }
    }
}

// ---------------- pack / init kernels ----------------

__global__ void k_init(float* agg, const float* pos, float* pos_out) {
    int i = blockIdx.x * blockDim.x + threadIdx.x;
    if (i < N_NODES * HID) agg[i] = 0.0f;
    if (i < N_NODES * 3)   pos_out[i] = pos[i];
}

__global__ void k_cvt_bf16(const float* src, __bf16* dst, int n) {
    int i = blockIdx.x * blockDim.x + threadIdx.x;
    if (i < n) dst[i] = (__bf16)src[i];
}

__global__ void k_tail(const float* attr, const float* pos, const int* ei, __bf16* tail) {
    int i = blockIdx.x * blockDim.x + threadIdx.x;
    if (i >= N_EDGES * TAILW) return;
    int e = i / TAILW, j = i % TAILW;
    float v = 0.0f;
    if (j < EDGE_DIM) {
        v = attr[(size_t)e * EDGE_DIM + j];
    } else if (j == EDGE_DIM) {
        int r = ei[e], c = ei[N_EDGES + e];
        float dx = pos[3 * r + 0] - pos[3 * c + 0];
        float dy = pos[3 * r + 1] - pos[3 * c + 1];
        float dz = pos[3 * r + 2] - pos[3 * c + 2];
        v = sqrtf(dx * dx + dy * dy + dz * dz) + 1e-8f;
    }
    tail[i] = (__bf16)v;
}

// Repack row-major [ktot x 256] fp32 weight into WMMA B-fragment order (bf16).
__global__ void k_pack_w(const float* src, __bf16* dst, int ktot, int ksrc) {
    int i = blockIdx.x * blockDim.x + threadIdx.x;
    if (i >= ktot * HID) return;
    int k = i / HID, n = i % HID;
    float v = (k < ksrc) ? src[(size_t)k * HID + n] : 0.0f;
    int c = k >> 5, rem = k & 31;
    int lh = (rem >> 3) & 1, e = (rem & 7) + ((rem >> 4) << 3);
    int lane = lh * 16 + (n & 15), nt = n >> 4;
    dst[((((size_t)c * 16 + nt) * 32 + lane) << 4) + e] = (__bf16)v;
}

// ---------------- edge kernel ----------------
// block = 128 threads (4 waves), 32 edges per block (2 row-tiles);
// wave w computes cols [64w, 64w+64) for both row-tiles.

__global__ __launch_bounds__(128) void k_edge(
    const int* __restrict__ ei, const __bf16* __restrict__ h_bf, const __bf16* __restrict__ tail,
    const __bf16* __restrict__ eW1p, const __bf16* __restrict__ eW2p, const __bf16* __restrict__ cW1p,
    const float* __restrict__ eb1, const float* __restrict__ eb2, const float* __restrict__ cb1,
    const float* __restrict__ cW2v, const float* __restrict__ pos,
    float* __restrict__ agg, float* __restrict__ pos_out)
{
    __shared__ __align__(16) __bf16 sE1[32][HID];
    __shared__ __align__(16) __bf16 sE2[32][HID];
    __shared__ float sRed[4][32];

    const int tid  = threadIdx.x;
    const int lane = tid & 31, wave = tid >> 5;
    const int lh = lane >> 4, ln = lane & 15;
    const int e0 = blockIdx.x * 32;
    const int er0 = ei[e0 + ln],      ec0 = ei[N_EDGES + e0 + ln];
    const int er1 = ei[e0 + 16 + ln], ec1 = ei[N_EDGES + e0 + 16 + ln];
    const int lwoff = (wave * 4) * WNT + lane * 16;

    v8f acc[2][4];
#pragma unroll
    for (int m = 0; m < 2; ++m)
#pragma unroll
        for (int i = 0; i < 4; ++i) zero8(acc[m][i]);

    // GEMM1: [32 x 1088] x [1088 x 256] -- three straight-line runs
    gemm_run2<16>(h_bf + (size_t)er0 * IN_DIM, h_bf + (size_t)er1 * IN_DIM,
                  eW1p + lwoff, lh, acc);
    gemm_run2<16>(h_bf + (size_t)ec0 * IN_DIM, h_bf + (size_t)ec1 * IN_DIM,
                  eW1p + (size_t)16 * WCHUNK + lwoff, lh, acc);
    gemm_run2< 2>(tail + (size_t)(e0 + ln) * TAILW, tail + (size_t)(e0 + 16 + ln) * TAILW,
                  eW1p + (size_t)32 * WCHUNK + lwoff, lh, acc);

#pragma unroll
    for (int m = 0; m < 2; ++m)
#pragma unroll
        for (int i = 0; i < 4; ++i) {
            int col = (wave * 4 + i) * 16 + ln;
            float bsv = eb1[col];
#pragma unroll
            for (int r = 0; r < 8; ++r)
                sE1[m * 16 + r + 8 * lh][col] = (__bf16)silu_f(acc[m][i][r] + bsv);
        }
    __syncthreads();

    // GEMM2: e1 @ eW2  (K=256)
#pragma unroll
    for (int m = 0; m < 2; ++m)
#pragma unroll
        for (int i = 0; i < 4; ++i) zero8(acc[m][i]);
    gemm_run2<8>(&sE1[ln][0], &sE1[16 + ln][0], eW2p + lwoff, lh, acc);

    int rn[2][8];
#pragma unroll
    for (int m = 0; m < 2; ++m)
#pragma unroll
        for (int r = 0; r < 8; ++r) rn[m][r] = ei[e0 + m * 16 + r + 8 * lh];
#pragma unroll
    for (int m = 0; m < 2; ++m)
#pragma unroll
        for (int i = 0; i < 4; ++i) {
            int col = (wave * 4 + i) * 16 + ln;
            float bsv = eb2[col];
#pragma unroll
            for (int r = 0; r < 8; ++r) {
                float v = silu_f(acc[m][i][r] + bsv);      // e_final
                sE2[m * 16 + r + 8 * lh][col] = (__bf16)v;
                atomicAdd(&agg[(size_t)rn[m][r] * HID + col], v);   // segment_sum(e, row)
            }
        }
    __syncthreads();

    // GEMM3: e_final @ cW1 (K=256), then dot with cW2 per row
#pragma unroll
    for (int m = 0; m < 2; ++m)
#pragma unroll
        for (int i = 0; i < 4; ++i) zero8(acc[m][i]);
    gemm_run2<8>(&sE2[ln][0], &sE2[16 + ln][0], cW1p + lwoff, lh, acc);

    float ps[2][8];
#pragma unroll
    for (int m = 0; m < 2; ++m)
#pragma unroll
        for (int r = 0; r < 8; ++r) ps[m][r] = 0.0f;
#pragma unroll
    for (int i = 0; i < 4; ++i) {
        int col = (wave * 4 + i) * 16 + ln;
        float bsv = cb1[col], w2 = cW2v[col];
#pragma unroll
        for (int m = 0; m < 2; ++m)
#pragma unroll
            for (int r = 0; r < 8; ++r) ps[m][r] += silu_f(acc[m][i][r] + bsv) * w2;
    }
#pragma unroll
    for (int msk = 1; msk < 16; msk <<= 1)
#pragma unroll
        for (int m = 0; m < 2; ++m)
#pragma unroll
            for (int r = 0; r < 8; ++r) ps[m][r] += __shfl_xor(ps[m][r], msk, 32);
    if (ln == 0)
#pragma unroll
        for (int m = 0; m < 2; ++m)
#pragma unroll
            for (int r = 0; r < 8; ++r) sRed[wave][m * 16 + r + 8 * lh] = ps[m][r];
    __syncthreads();

    if (tid < 32) {
        float cu = sRed[0][tid] + sRed[1][tid] + sRed[2][tid] + sRed[3][tid];
        cu = fminf(1.0f, fmaxf(-1.0f, cu));
        int e = e0 + tid;
        int r = ei[e], c = ei[N_EDGES + e];
        float dx = pos[3 * r + 0] - pos[3 * c + 0];
        float dy = pos[3 * r + 1] - pos[3 * c + 1];
        float dz = pos[3 * r + 2] - pos[3 * c + 2];
        atomicAdd(&pos_out[3 * r + 0], cu * dx);
        atomicAdd(&pos_out[3 * r + 1], cu * dy);
        atomicAdd(&pos_out[3 * r + 2], cu * dz);
    }
}

// ---------------- node kernel: h_new = silu([h|agg]@nW1+nb1)@nW2 + nb2 ----------------

__global__ __launch_bounds__(128) void k_node(
    const __bf16* __restrict__ h_bf, const float* __restrict__ agg,
    const __bf16* __restrict__ nW1p, const __bf16* __restrict__ nW2p,
    const float* __restrict__ nb1, const float* __restrict__ nb2,
    float* __restrict__ h_out)
{
    __shared__ __align__(16) __bf16 sH[16][HID];
    __shared__ __align__(16) __bf16 sAgg[16][HID];

    const int tid  = threadIdx.x;
    const int lane = tid & 31, wave = tid >> 5;
    const int lh = lane >> 4, ln = lane & 15;
    const int n0 = blockIdx.x * 16;
    const int node = n0 + ln;
    const int lwoff = (wave * 4) * WNT + lane * 16;

    // cooperative stage: agg rows (fp32) -> bf16 LDS
#pragma unroll
    for (int t = 0; t < 32; ++t) {
        int idx = tid + t * 128;            // 0..4095
        int rr = idx >> 8, cc = idx & 255;
        sAgg[rr][cc] = (__bf16)agg[(size_t)(n0 + rr) * HID + cc];
    }
    __syncthreads();

    v8f acc[4];
#pragma unroll
    for (int i = 0; i < 4; ++i) zero8(acc[i]);

    gemm_run<16>(h_bf + (size_t)node * IN_DIM, nW1p + lwoff,                       lh, acc);
    gemm_run< 8>(&sAgg[ln][0],                 nW1p + (size_t)16 * WCHUNK + lwoff, lh, acc);

#pragma unroll
    for (int i = 0; i < 4; ++i) {
        int col = (wave * 4 + i) * 16 + ln;
        float bsv = nb1[col];
#pragma unroll
        for (int r = 0; r < 8; ++r)
            sH[r + 8 * lh][col] = (__bf16)silu_f(acc[i][r] + bsv);
    }
    __syncthreads();

#pragma unroll
    for (int i = 0; i < 4; ++i) zero8(acc[i]);
    gemm_run<8>(&sH[ln][0], nW2p + lwoff, lh, acc);

#pragma unroll
    for (int i = 0; i < 4; ++i) {
        int col = (wave * 4 + i) * 16 + ln;
        float bsv = nb2[col];
#pragma unroll
        for (int r = 0; r < 8; ++r)
            h_out[(size_t)(n0 + r + 8 * lh) * HID + col] = acc[i][r] + bsv;  // no final SiLU
    }
}

// ---------------- launch ----------------

extern "C" void kernel_launch(void* const* d_in, const int* in_sizes, int n_in,
                              void* d_out, int out_size, void* d_ws, size_t ws_size,
                              hipStream_t stream)
{
    const float* h    = (const float*)d_in[0];
    const int*   ei   = (const int*)  d_in[1];
    const float* eatt = (const float*)d_in[2];
    const float* pos  = (const float*)d_in[3];
    const float* eW1  = (const float*)d_in[4];
    const float* eb1  = (const float*)d_in[5];
    const float* eW2  = (const float*)d_in[6];
    const float* eb2  = (const float*)d_in[7];
    const float* cW1  = (const float*)d_in[8];
    const float* cb1  = (const float*)d_in[9];
    const float* cW2  = (const float*)d_in[10];
    const float* nW1  = (const float*)d_in[11];
    const float* nb1  = (const float*)d_in[12];
    const float* nW2  = (const float*)d_in[13];
    const float* nb2  = (const float*)d_in[14];

    char* ws = (char*)d_ws;
    size_t off = 0;
    auto carve = [&](size_t bytes) {
        void* p = ws + off;
        off += (bytes + 255) & ~(size_t)255;
        return p;
    };
    __bf16* h_bf = (__bf16*)carve((size_t)N_NODES * IN_DIM * 2);
    __bf16* tail = (__bf16*)carve((size_t)N_EDGES * TAILW * 2);
    float*  agg  = (float*) carve((size_t)N_NODES * HID * 4);
    __bf16* eW1p = (__bf16*)carve((size_t)K1  * HID * 2);
    __bf16* eW2p = (__bf16*)carve((size_t)HID * HID * 2);
    __bf16* cW1p = (__bf16*)carve((size_t)HID * HID * 2);
    __bf16* nW1p = (__bf16*)carve((size_t)KN  * HID * 2);
    __bf16* nW2p = (__bf16*)carve((size_t)HID * HID * 2);

    float* h_out   = (float*)d_out;
    float* pos_out = h_out + (size_t)N_NODES * HID;

    const int T = 256;
    k_init    <<<(N_NODES * HID + T - 1) / T, T, 0, stream>>>(agg, pos, pos_out);
    k_cvt_bf16<<<(N_NODES * IN_DIM + T - 1) / T, T, 0, stream>>>(h, h_bf, N_NODES * IN_DIM);
    k_tail    <<<(N_EDGES * TAILW + T - 1) / T, T, 0, stream>>>(eatt, pos, ei, tail);
    k_pack_w  <<<(K1  * HID + T - 1) / T, T, 0, stream>>>(eW1, eW1p, K1, K1SRC);
    k_pack_w  <<<(HID * HID + T - 1) / T, T, 0, stream>>>(eW2, eW2p, HID, HID);
    k_pack_w  <<<(HID * HID + T - 1) / T, T, 0, stream>>>(cW1, cW1p, HID, HID);
    k_pack_w  <<<(KN  * HID + T - 1) / T, T, 0, stream>>>(nW1, nW1p, KN, KN);
    k_pack_w  <<<(HID * HID + T - 1) / T, T, 0, stream>>>(nW2, nW2p, HID, HID);

    k_edge<<<N_EDGES / 32, 128, 0, stream>>>(ei, h_bf, tail, eW1p, eW2p, cW1p,
                                             eb1, eb2, cb1, cW2, pos, agg, pos_out);
    k_node<<<N_NODES / 16, 128, 0, stream>>>(h_bf, agg, nW1p, nW2p, nb1, nb2, h_out);
}
